// GroupQueryAttention_81157702025389
// MI455X (gfx1250) — matrix operand order repaired and checked by reading, not compiled, not run on hardware
//
#include <hip/hip_runtime.h>
#include <hip/hip_bf16.h>
#include <math.h>

typedef __bf16 bf16;
typedef __attribute__((ext_vector_type(16))) __bf16 v16bf;
typedef __attribute__((ext_vector_type(8)))  __bf16 v8bf;
typedef __attribute__((ext_vector_type(4)))  __bf16 v4bf;
typedef __attribute__((ext_vector_type(2)))  __bf16 v2bf;
typedef __attribute__((ext_vector_type(8)))  float  v8f;
typedef __attribute__((ext_vector_type(4)))  float  v4f;

#define B_    2
#define T_    2048
#define DIN_  1024
#define DOUT_ 1024
#define G_    4
#define H_    16
#define HPG_  4
#define DH_   64
#define NEG_BIG (-1.0e30f)

// ---------------------------------------------------------------------------
// CDNA5 wave32 WMMA fragment layouts (ISA 05_wmma §7.12.2), bf16 16x16x32:
//  A: lane<16 row M=lane, K={0..7,16..23}; lane>=16 row M=lane-16, K={8..15,24..31}
//  B: lane col N=lane&15; lane<16 K={0..7,16..23}; lane>=16 K={8..15,24..31} (same as A)
//  C/D: elem r -> (M = r + (lane&16?8:0), N = lane&15)
// ---------------------------------------------------------------------------

__device__ __forceinline__ v8f v8f_zero() {
  v8f z;
#pragma unroll
  for (int i = 0; i < 8; ++i) z[i] = 0.0f;
  return z;
}

__device__ __forceinline__ v16bf frag_pair(const bf16* p0, const bf16* p1) {
  v8bf lo = *(const v8bf*)p0;          // 16B each, must be 16B aligned
  v8bf hi = *(const v8bf*)p1;
  v16bf r;
#pragma unroll
  for (int i = 0; i < 8; ++i) { r[i] = lo[i]; r[8 + i] = hi[i]; }
  return r;
}

__device__ __forceinline__ v16bf frag_a(const bf16* row, int lane) {
  const int kb = (lane & 16) ? 8 : 0;
  return frag_pair(row + kb, row + kb + 16);
}

__device__ __forceinline__ v16bf frag_b(const bf16* nrow, int lane) {
  const int kb = (lane & 16) ? 8 : 0;
  return frag_pair(nrow + kb, nrow + kb + 16);
}

// CDNA5 async LDS DMA: each active lane copies 16B global -> LDS (ASYNCcnt).
__device__ __forceinline__ void async_copy_b128(const bf16* g, bf16* l) {
  unsigned lds_addr = (unsigned)(size_t)l;   // low 32 bits = LDS byte offset
  asm volatile("global_load_async_to_lds_b128 %0, %1, off"
               :: "v"(lds_addr), "v"(g) : "memory");
}
__device__ __forceinline__ void wait_asynccnt0() {
  asm volatile("s_wait_asynccnt 0x0" ::: "memory");
}

// ---------------------------------------------------------------------------
// GEMM: C[M,N] = A[M,K] @ B[K,N]; fp32 in/out, bf16 WMMA compute.
// 128x64 tile, BK=32, 256 threads = 8 waves, 4 WMMA per wave per K-step.
// ---------------------------------------------------------------------------
#define BM 128
#define BN 64
#define BK 32
#define LDA  48   // bf16 elems; 96B rows: 16B aligned, bank-phase skewed
#define LDBT 48

__global__ __launch_bounds__(256) void gemm_f32_bf16wmma(
    const float* __restrict__ A, const float* __restrict__ Bm,
    float* __restrict__ C, int M, int N, int K)
{
  __shared__ __attribute__((aligned(16))) bf16 As[BM * LDA];
  __shared__ __attribute__((aligned(16))) bf16 BsT[BN * LDBT];
  const int tid  = threadIdx.x;
  const int lane = tid & 31;
  const int wave = tid >> 5;
  const int bm = blockIdx.y * BM;
  const int bn = blockIdx.x * BN;
  const int mr = wave * 16;                 // wave's 16-row band

  v8f acc[4];
#pragma unroll
  for (int j = 0; j < 4; ++j) acc[j] = v8f_zero();

  const int pr = tid >> 4;                  // B: K-row pair 0..15
  const int pc = (tid & 15) * 4;            // B: col group

  for (int k0 = 0; k0 < K; k0 += BK) {
    // ---- batch ALL global loads first (loads stay in flight together) ----
    v4f aL[4];
    int ar[4], ac[4];
#pragma unroll
    for (int i = 0; i < 4; ++i) {           // A tile 128x32
      int idx = (i * 256 + tid) * 4;
      ar[i] = idx >> 5; ac[i] = idx & 31;
      aL[i] = *(const v4f*)(A + (size_t)(bm + ar[i]) * K + k0 + ac[i]);
    }
    v4f b0 = *(const v4f*)(Bm + (size_t)(k0 + 2 * pr) * N + bn + pc);
    v4f b1 = *(const v4f*)(Bm + (size_t)(k0 + 2 * pr + 1) * N + bn + pc);

    // ---- convert + store to LDS ----
#pragma unroll
    for (int i = 0; i < 4; ++i) {
      v4bf ab;
#pragma unroll
      for (int e = 0; e < 4; ++e) ab[e] = (bf16)aL[i][e];
      *(v4bf*)&As[ar[i] * LDA + ac[i]] = ab;
    }
#pragma unroll
    for (int e = 0; e < 4; ++e) {           // transposed, packed pair store
      v2bf pk; pk[0] = (bf16)b0[e]; pk[1] = (bf16)b1[e];
      *(v2bf*)&BsT[(pc + e) * LDBT + 2 * pr] = pk;
    }
    if (k0 + BK < K) {                      // prefetch next K-tile
      __builtin_prefetch(A + (size_t)(bm + (tid >> 1)) * K + k0 + BK + (tid & 1) * 16, 0, 1);
      __builtin_prefetch(Bm + (size_t)(k0 + BK + (tid >> 3)) * N + bn + (tid & 7) * 8, 0, 1);
    }
    __syncthreads();

    v16bf af = frag_a(&As[(mr + (lane & 15)) * LDA], lane);
#pragma unroll
    for (int j = 0; j < 4; ++j) {
      v16bf bfj = frag_b(&BsT[(j * 16 + (lane & 15)) * LDBT], lane);
      acc[j] = __builtin_amdgcn_wmma_f32_16x16x32_bf16(false, af, false, bfj,
                                                       (short)0, acc[j], false, false);
    }
    __syncthreads();
  }

  const int rb  = (lane & 16) ? 8 : 0;
  const int col = lane & 15;
#pragma unroll
  for (int j = 0; j < 4; ++j)
#pragma unroll
    for (int r = 0; r < 8; ++r)
      C[(size_t)(bm + mr + rb + r) * N + bn + j * 16 + col] = acc[j][r];
}

// ---------------------------------------------------------------------------
// RoPE + bf16 repack:
//  Q [B,T,1024] -> qb [B,G,Hpg,T,DH];  K,V [B,T,256] -> kb/vb [B,G,T,DH]
// ---------------------------------------------------------------------------
__global__ __launch_bounds__(256) void rope_pack(
    const float* __restrict__ qf, const float* __restrict__ kf,
    const float* __restrict__ vf, const float* __restrict__ cosT,
    const float* __restrict__ sinT,
    bf16* __restrict__ qb, bf16* __restrict__ kb, bf16* __restrict__ vb)
{
  const int bt = blockIdx.x;               // 0 .. B*T-1
  const int b  = bt / T_;
  const int t  = bt % T_;
  const int tid = threadIdx.x;
  const float* cr = cosT + (size_t)t * DH_;
  const float* sr = sinT + (size_t)t * DH_;

#pragma unroll
  for (int j = 0; j < 4; ++j) {            // Q: 1024 elems / 256 threads
    int d = j * 256 + tid;
    int g = d >> 8, rem = d & 255, hpg = rem >> 6, i = rem & 63;
    float v  = qf[(size_t)bt * DOUT_ + d];
    float vr = (i < 32) ? -qf[(size_t)bt * DOUT_ + d + 32]
                        :  qf[(size_t)bt * DOUT_ + d - 32];
    qb[((((size_t)b * G_ + g) * HPG_ + hpg) * T_ + t) * DH_ + i] =
        (bf16)(v * cr[i] + vr * sr[i]);
  }
  {                                        // K + V: 256 elems each
    int d = tid;
    int g = d >> 6, i = d & 63;
    float v  = kf[(size_t)bt * (G_ * DH_) + d];
    float vr = (i < 32) ? -kf[(size_t)bt * (G_ * DH_) + d + 32]
                        :  kf[(size_t)bt * (G_ * DH_) + d - 32];
    kb[(((size_t)b * G_ + g) * T_ + t) * DH_ + i] = (bf16)(v * cr[i] + vr * sr[i]);
    vb[(((size_t)b * G_ + g) * T_ + t) * DH_ + i] =
        (bf16)vf[(size_t)bt * (G_ * DH_) + d];
  }
}

// ---------------------------------------------------------------------------
// Flash-style causal attention. 4 waves/block, 64 query rows per block (16
// per wave). K tile staged via CDNA5 async LDS DMA; V tile transposed with
// packed pair stores; online softmax with vectorized LDS and shfl combine.
// Scale = 1/sqrt(T) (reference divides by sqrt(token_len)).
// ---------------------------------------------------------------------------
#define QT 64
#define KT 64
#define NW 4
#define LDK  80   // 160B rows: 16B aligned
#define LDVT 80
#define LDS_S 68  // 272B rows: 16B aligned

__global__ __launch_bounds__(128) void attn_flash(
    const bf16* __restrict__ qb, const bf16* __restrict__ kb,
    const bf16* __restrict__ vb, float* __restrict__ attn /*[B,T,DOUT]*/)
{
  __shared__ __attribute__((aligned(16))) bf16 Ks[KT * LDK];    // [key][dh]
  __shared__ __attribute__((aligned(16))) bf16 VsT[DH_ * LDVT]; // [dh][key]
  __shared__ __attribute__((aligned(16))) float Ssm[NW][16][LDS_S];
  __shared__ __attribute__((aligned(16))) bf16 Psm[NW][16][KT];
  __shared__ float Alp[NW][16];
  __shared__ float Linv[NW][16];

  const int tid  = threadIdx.x;
  const int lane = tid & 31;
  const int wave = tid >> 5;
  const int qt0  = blockIdx.x * QT;
  const int head = blockIdx.y;             // 0 .. B*G*Hpg-1
  const int b    = head / (G_ * HPG_);
  const int g    = (head / HPG_) % G_;
  const int hpg  = head % HPG_;

  const bf16* qh = qb + (((size_t)b * G_ + g) * HPG_ + hpg) * T_ * DH_;
  const bf16* kh = kb + ((size_t)b * G_ + g) * T_ * DH_;
  const bf16* vh = vb + ((size_t)b * G_ + g) * T_ * DH_;

  const int   qbase = qt0 + wave * 16;
  const float scale = rsqrtf((float)T_);

  v16bf qfr[2];                            // Q fragment, loop-invariant
#pragma unroll
  for (int ks = 0; ks < 2; ++ks)
    qfr[ks] = frag_a(qh + (size_t)(qbase + (lane & 15)) * DH_ + ks * 32, lane);

  v8f O[4];
#pragma unroll
  for (int j = 0; j < 4; ++j) O[j] = v8f_zero();

  float m_i = NEG_BIG, l_i = 0.0f;

  const int nkt = qt0 / KT + 1;            // uniform across the block
  for (int kt = 0; kt < nkt; ++kt) {
    const int s0 = kt * KT;

    // ---- K tile: async LDS DMA (pure bf16 copy, 16B per lane per issue) ----
#pragma unroll
    for (int i = 0; i < 4; ++i) {
      int idx = (i * 128 + tid) * 8;       // 4096 bf16
      int kk = idx >> 6, dd = idx & 63;
      async_copy_b128(kh + (size_t)(s0 + kk) * DH_ + dd, &Ks[kk * LDK + dd]);
    }
    // ---- V tile: batched loads, transposed packed-pair stores ----
    v8bf vL[2][2];
    int kp[2], dg[2];
#pragma unroll
    for (int i = 0; i < 2; ++i) {
      int c = i * 128 + tid;               // 0..255 chunks
      kp[i] = c >> 3; dg[i] = (c & 7) * 8;
      vL[i][0] = *(const v8bf*)(vh + (size_t)(s0 + 2 * kp[i]) * DH_ + dg[i]);
      vL[i][1] = *(const v8bf*)(vh + (size_t)(s0 + 2 * kp[i] + 1) * DH_ + dg[i]);
    }
#pragma unroll
    for (int i = 0; i < 2; ++i)
#pragma unroll
      for (int e = 0; e < 8; ++e) {
        v2bf pk; pk[0] = vL[i][0][e]; pk[1] = vL[i][1][e];
        *(v2bf*)&VsT[(dg[i] + e) * LDVT + 2 * kp[i]] = pk;
      }
    if (kt + 1 < nkt) {                    // prefetch next KV tile
      __builtin_prefetch(kh + (size_t)(s0 + KT + (tid >> 1)) * DH_ + (tid & 1) * 32, 0, 1);
      __builtin_prefetch(vh + (size_t)(s0 + KT + (tid >> 1)) * DH_ + (tid & 1) * 32, 0, 1);
    }
    wait_asynccnt0();                      // K DMA landed in LDS
    __syncthreads();

    // ---- scores S = scale * Q K^T, causal-masked, staged to LDS ----
#pragma unroll
    for (int j = 0; j < 4; ++j) {
      v8f s = v8f_zero();
#pragma unroll
      for (int ks = 0; ks < 2; ++ks) {
        v16bf kfr = frag_b(&Ks[(j * 16 + (lane & 15)) * LDK + ks * 32], lane);
        s = __builtin_amdgcn_wmma_f32_16x16x32_bf16(false, qfr[ks], false, kfr,
                                                    (short)0, s, false, false);
      }
      const int rb  = (lane & 16) ? 8 : 0;
      const int col = s0 + j * 16 + (lane & 15);
#pragma unroll
      for (int r = 0; r < 8; ++r) {
        int   row = qbase + rb + r;
        float v   = s[r] * scale;
        if (col > row) v = NEG_BIG;        // causal mask
        Ssm[wave][rb + r][j * 16 + (lane & 15)] = v;
      }
    }
    __syncthreads();

    // ---- online softmax: 2 lanes per row, vectorized LDS, shfl combine ----
    {
      const int row = lane & 15;
      const int c0  = (lane >> 4) * 32;
      float mt = NEG_BIG;
#pragma unroll
      for (int c = 0; c < 8; ++c) {
        v4f sv = *(const v4f*)&Ssm[wave][row][c0 + c * 4];
        mt = fmaxf(mt, fmaxf(fmaxf(sv[0], sv[1]), fmaxf(sv[2], sv[3])));
      }
      mt = fmaxf(mt, __shfl_xor(mt, 16, 32));
      float mn = fmaxf(m_i, mt);
      float al = __expf(m_i - mn);
      float ps = 0.0f;
#pragma unroll
      for (int c = 0; c < 8; ++c) {
        v4f sv = *(const v4f*)&Ssm[wave][row][c0 + c * 4];
        v4bf pb;
#pragma unroll
        for (int e = 0; e < 4; ++e) {
          float p = __expf(sv[e] - mn);
          ps += p;
          pb[e] = (bf16)p;
        }
        *(v4bf*)&Psm[wave][row][c0 + c * 4] = pb;
      }
      ps += __shfl_xor(ps, 16, 32);
      l_i = l_i * al + ps;
      m_i = mn;
      if (lane < 16) Alp[wave][lane] = al;
    }
    __syncthreads();

    // ---- rescale O by alpha(row), then O += P @ V via WMMA ----
    const int rb = (lane & 16) ? 8 : 0;
    float alv[8];
#pragma unroll
    for (int r = 0; r < 8; ++r) alv[r] = Alp[wave][rb + r];
#pragma unroll
    for (int j = 0; j < 4; ++j)
#pragma unroll
      for (int r = 0; r < 8; ++r) O[j][r] *= alv[r];

#pragma unroll
    for (int ks = 0; ks < 2; ++ks) {
      v16bf pfr = frag_a(&Psm[wave][lane & 15][ks * 32], lane);
#pragma unroll
      for (int j = 0; j < 4; ++j) {
        v16bf vfr = frag_b(&VsT[(j * 16 + (lane & 15)) * LDVT + ks * 32], lane);
        O[j] = __builtin_amdgcn_wmma_f32_16x16x32_bf16(false, pfr, false, vfr,
                                                       (short)0, O[j], false, false);
      }
    }
    __syncthreads();                        // protect Ks/VsT for next tile
  }

  if (lane < 16) Linv[wave][lane] = 1.0f / l_i;
  __syncthreads();

  const int rb   = (lane & 16) ? 8 : 0;
  const int dcol = lane & 15;
#pragma unroll
  for (int j = 0; j < 4; ++j)
#pragma unroll
    for (int r = 0; r < 8; ++r) {
      int row = qbase + rb + r;
      int dh  = j * 16 + dcol;
      attn[((size_t)b * T_ + row) * DOUT_ + g * (HPG_ * DH_) + hpg * DH_ + dh] =
          O[j][r] * Linv[wave][rb + r];
    }
}

// ---------------------------------------------------------------------------
// Host-side orchestration (graph-capture safe: launches only, on `stream`)
// ---------------------------------------------------------------------------
#define MB (1024ull * 1024ull)

extern "C" void kernel_launch(void* const* d_in, const int* in_sizes, int n_in,
                              void* d_out, int out_size, void* d_ws, size_t ws_size,
                              hipStream_t stream) {
  (void)in_sizes; (void)n_in; (void)out_size; (void)ws_size;
  const float* x    = (const float*)d_in[0];
  const float* Wq   = (const float*)d_in[1];
  const float* Wk   = (const float*)d_in[2];
  const float* Wv   = (const float*)d_in[3];
  const float* Wo   = (const float*)d_in[4];
  const float* cosT = (const float*)d_in[5];
  const float* sinT = (const float*)d_in[6];
  float* out = (float*)d_out;

  char* ws = (char*)d_ws;
  float* q_f    = (float*)(ws + 0 * MB);    // [B*T, 1024]    16 MB
  float* k_f    = (float*)(ws + 16 * MB);   // [B*T, 256]      4 MB
  float* v_f    = (float*)(ws + 20 * MB);   // [B*T, 256]      4 MB
  float* attn_f = (float*)(ws + 24 * MB);   // [B,T,1024]     16 MB
  bf16*  q_bf   = (bf16*)(ws + 40 * MB);    // [B,G,Hpg,T,DH]  8 MB
  bf16*  k_bf   = (bf16*)(ws + 48 * MB);    // [B,G,T,DH]      2 MB
  bf16*  v_bf   = (bf16*)(ws + 50 * MB);    // [B,G,T,DH]      2 MB

  const int M = B_ * T_;                    // 4096

  gemm_f32_bf16wmma<<<dim3(DOUT_ / BN, M / BM), 256, 0, stream>>>(
      x, Wq, q_f, M, DOUT_, DIN_);
  gemm_f32_bf16wmma<<<dim3((G_ * DH_) / BN, M / BM), 256, 0, stream>>>(
      x, Wk, k_f, M, G_ * DH_, DIN_);
  gemm_f32_bf16wmma<<<dim3((G_ * DH_) / BN, M / BM), 256, 0, stream>>>(
      x, Wv, v_f, M, G_ * DH_, DIN_);

  rope_pack<<<dim3(B_ * T_), 256, 0, stream>>>(q_f, k_f, v_f, cosT, sinT,
                                               q_bf, k_bf, v_bf);

  attn_flash<<<dim3(T_ / QT, B_ * G_ * HPG_), 128, 0, stream>>>(
      q_bf, k_bf, v_bf, attn_f);

  gemm_f32_bf16wmma<<<dim3(DOUT_ / BN, M / BM), 256, 0, stream>>>(
      attn_f, Wo, out, M, DOUT_, DIN_);
}
